// GraphEncoder_12867722018904
// MI455X (gfx1250) — matrix operand order repaired and checked by reading, not compiled, not run on hardware
//
#include <hip/hip_runtime.h>

typedef __attribute__((ext_vector_type(16))) __bf16 v16bf_t;
typedef __attribute__((ext_vector_type(8)))  __bf16 v8bf_t;
typedef __attribute__((ext_vector_type(8)))  float  v8f_t;

constexpr int kNPG     = 9;     // nodes per graph
constexpr int kRank    = 128;
constexpr int kOrder   = 3;
constexpr int kEPG     = 25;    // 16 random edges + 9 self loops, contiguous per graph
constexpr int kGPB     = 16;    // graphs per block -> 144 nodes = 9 row-tiles of 16
constexpr int kNodes   = kGPB * kNPG;   // 144
constexpr int kThreads = 128;           // 4 wave32
constexpr int kSH = kRank + 4;          // f32 row stride (pad -> 4-bank row skew)
constexpr int kSA = kRank + 8;          // bf16 row stride: 272B (16B aligned, bank skew)
constexpr int kSB = kRank + 8;

// Dynamic LDS layout (bytes). fp32 arrays first, then 16B-aligned bf16 tiles.
constexpr size_t kOffH    = 0;                                              // feats / GEMM-out (aliased)
constexpr size_t kOffNs   = kOffH    + (size_t)kNodes * kSH * sizeof(float);
constexpr size_t kOffNd   = kOffNs   + (size_t)kNodes * sizeof(float);
constexpr size_t kOffCoef = kOffNd   + (size_t)kNodes * sizeof(float);
constexpr size_t kOffA    = kOffCoef + (size_t)kGPB * 81 * sizeof(float);   // 82368, 16B aligned
constexpr size_t kOffB    = kOffA    + (size_t)kNodes * kSA * 2;            // bf16 A tile
constexpr size_t kOffSrc  = kOffB    + (size_t)kRank  * kSB * 2;            // bf16 W^T tile
constexpr size_t kOffDst  = kOffSrc  + (size_t)kGPB * kEPG * sizeof(short);
constexpr size_t kSmem    = kOffDst  + (size_t)kGPB * kEPG * sizeof(short); // ~158 KB < 320 KB
static_assert(kOffA % 16 == 0, "A tile must be 16B aligned");
static_assert(kSmem < 320u * 1024u, "fits one WGP LDS");

__device__ inline v16bf_t bf16cat(v8bf_t lo, v8bf_t hi) {
  v16bf_t r;
#pragma unroll
  for (int i = 0; i < 8; ++i) { r[i] = lo[i]; r[i + 8] = hi[i]; }
  return r;
}

__global__ __launch_bounds__(kThreads)
void gcn_fused_kernel(const int* __restrict__ features,
                      const int* __restrict__ edge_src,
                      const int* __restrict__ edge_dst,
                      const float* __restrict__ emb,
                      const float* __restrict__ W,
                      const float* __restrict__ bias,
                      const float* __restrict__ gamma,
                      const float* __restrict__ beta,
                      float* __restrict__ out, int nGraphs) {
  extern __shared__ __align__(16) unsigned char smem[];
  float*  sH    = reinterpret_cast<float*>(smem + kOffH);     // [144][132] fp32
  float*  sNs   = reinterpret_cast<float*>(smem + kOffNs);    // [144]
  float*  sNd   = reinterpret_cast<float*>(smem + kOffNd);    // [144]
  float*  sCoef = reinterpret_cast<float*>(smem + kOffCoef);  // [16][9][9]
  __bf16* sA    = reinterpret_cast<__bf16*>(smem + kOffA);    // [144][136] bf16
  __bf16* sB    = reinterpret_cast<__bf16*>(smem + kOffB);    // [128][136] bf16 (W^T)
  short*  sSrc  = reinterpret_cast<short*>(smem + kOffSrc);   // [16][25] local ids
  short*  sDst  = reinterpret_cast<short*>(smem + kOffDst);

  const int tid = threadIdx.x;
  const int g0  = blockIdx.x * kGPB;

  // ---- edges -> LDS as graph-local node ids ----
  for (int i = tid; i < kGPB * kEPG; i += kThreads) {
    int g = i / kEPG, e = i - g * kEPG;
    int gg = g0 + g; if (gg >= nGraphs) gg = nGraphs - 1;
    long long ebase = (long long)gg * kEPG;
    sSrc[i] = (short)(edge_src[ebase + e] - gg * kNPG);
    sDst[i] = (short)(edge_dst[ebase + e] - gg * kNPG);
  }
  __syncthreads();

  // ---- degree -> rsqrt norms (self-loops guarantee deg >= 1, clip anyway) ----
  for (int v = tid; v < kNodes; v += kThreads) {
    int g = v / kNPG, l = v - g * kNPG;
    const short* es = &sSrc[g * kEPG];
    const short* ed = &sDst[g * kEPG];
    int dout = 0, din = 0;
#pragma unroll
    for (int e = 0; e < kEPG; ++e) { dout += (es[e] == l); din += (ed[e] == l); }
    if (dout < 1) dout = 1;
    if (din  < 1) din  = 1;
    sNs[v] = rsqrtf((float)dout);
    sNd[v] = rsqrtf((float)din);
  }
  __syncthreads();

  // ---- fold scatter-sum into a per-graph 9x9 coefficient matrix ----
  for (int i = tid; i < kGPB * 81; i += kThreads) {
    int g = i / 81, p = i - g * 81;
    int v = p / 9,  u = p - v * 9;
    const short* es = &sSrc[g * kEPG];
    const short* ed = &sDst[g * kEPG];
    int cnt = 0;
#pragma unroll
    for (int e = 0; e < kEPG; ++e) cnt += ((es[e] == u) & (ed[e] == v));
    sCoef[i] = (float)cnt * sNs[g * kNPG + u] * sNd[g * kNPG + v];
  }

  // ---- embedding gather: feats fp32 into sH ----
  for (int i = tid; i < kNodes * kRank; i += kThreads) {
    int v = i >> 7, f = i & (kRank - 1);
    int node = g0 * kNPG + v;
    int maxNode = nGraphs * kNPG - 1;
    if (node > maxNode) node = maxNode;
    int op = features[node];
    sH[v * kSH + f] = emb[op * kRank + f];
  }
  __syncthreads();

  const int wave = tid >> 5, lane = tid & 31;
  const int nh = lane >> 4;      // which K/M half this lane serves
  const int nl = lane & 15;

  for (int layer = 0; layer < kOrder; ++layer) {
    // Phase A: agg[v] = sum_u coef[v][u] * feats[u]  (per graph), -> bf16 A tile
    for (int i = tid; i < kNodes * kRank; i += kThreads) {
      int v = i >> 7, f = i & (kRank - 1);
      int g = v / kNPG, l = v - g * kNPG;
      const float* cf = &sCoef[(g * 9 + l) * 9];
      const float* fb = &sH[(g * kNPG) * kSH + f];
      float acc = 0.f;
#pragma unroll
      for (int u = 0; u < 9; ++u) acc = fmaf(cf[u], fb[u * kSH], acc);
      sA[v * kSA + f] = (__bf16)acc;
    }
    // Stage W[layer]^T as bf16 (column-contiguous for the B fragment layout)
    const float* Wl = W + (size_t)layer * kRank * kRank;
    for (int i = tid; i < kRank * kRank; i += kThreads) {
      int k = i >> 7, n = i & (kRank - 1);
      sB[n * kSB + k] = (__bf16)Wl[i];
    }
    __syncthreads();

    // Phase B: [144x128] @ [128x128] via v_wmma_f32_16x16x32_bf16.
    // Wave w owns column tiles {2w, 2w+1}; B fragments live in VGPRs.
    v16bf_t bfrag[2][4];
#pragma unroll
    for (int t = 0; t < 2; ++t) {
      const __bf16* bp = &sB[((wave * 2 + t) * 16 + nl) * kSB];
#pragma unroll
      for (int ks = 0; ks < 4; ++ks) {
        // lane holds column n, K = ks*32 + nh*16 + [0..15] (contiguous)
        const v8bf_t* p = reinterpret_cast<const v8bf_t*>(bp + ks * 32 + nh * 16);
        bfrag[t][ks] = bf16cat(p[0], p[1]);
      }
    }
    for (int mt = 0; mt < 9; ++mt) {
      v8f_t acc0 = {}, acc1 = {};
      const __bf16* ap = &sA[(mt * 16 + nl) * kSA];
#pragma unroll
      for (int ks = 0; ks < 4; ++ks) {
        // A layout: lane holds row, K = ks*32 + nh*16 + [0..15] (contiguous)
        const v8bf_t* p0 = reinterpret_cast<const v8bf_t*>(ap + ks * 32 + nh * 16);
        const v8bf_t* p1 = reinterpret_cast<const v8bf_t*>(ap + ks * 32 + nh * 16 + 8);
        v16bf_t afrag = bf16cat(p0[0], p1[0]);
        acc0 = __builtin_amdgcn_wmma_f32_16x16x32_bf16(false, afrag, false, bfrag[0][ks],
                                                       (short)0, acc0, false, false);
        acc1 = __builtin_amdgcn_wmma_f32_16x16x32_bf16(false, afrag, false, bfrag[1][ks],
                                                       (short)0, acc1, false, false);
      }
      // D layout: vgpr r, lanes 0-15 -> M=r, lanes 16-31 -> M=8+r, N=lane&15
      int mb = mt * 16 + nh * 8;
      float* h0 = &sH[mb * kSH + (wave * 2 + 0) * 16 + nl];
      float* h1 = &sH[mb * kSH + (wave * 2 + 1) * 16 + nl];
#pragma unroll
      for (int r = 0; r < 8; ++r) { h0[r * kSH] = acc0[r]; h1[r * kSH] = acc1[r]; }
    }
    __syncthreads();

    // Phase C: bias + LayerNorm + ReLU, in place on sH rows -> next feats
    const float* bl = bias  + layer * kRank;
    const float* gl = gamma + layer * kRank;
    const float* el = beta  + layer * kRank;
    for (int v = tid; v < kNodes; v += kThreads) {
      float* row = &sH[v * kSH];
      float mu = 0.f;
      for (int f = 0; f < kRank; ++f) mu += row[f] + bl[f];
      mu *= (1.f / kRank);
      float var = 0.f;
      for (int f = 0; f < kRank; ++f) { float x = row[f] + bl[f] - mu; var = fmaf(x, x, var); }
      float inv = rsqrtf(var * (1.f / kRank) + 1e-5f);
      for (int f = 0; f < kRank; ++f) {
        float x = (row[f] + bl[f] - mu) * inv * gl[f] + el[f];
        row[f] = x > 0.f ? x : 0.f;
      }
    }
    __syncthreads();
  }

  // ---- emit node 0 of every graph (coalesced) ----
  for (int i = tid; i < kGPB * kRank; i += kThreads) {
    int g = i >> 7, f = i & (kRank - 1);
    if (g0 + g < nGraphs)
      out[(size_t)(g0 + g) * kRank + f] = sH[(g * kNPG) * kSH + f];
  }
}

extern "C" void kernel_launch(void* const* d_in, const int* in_sizes, int n_in,
                              void* d_out, int out_size, void* d_ws, size_t ws_size,
                              hipStream_t stream) {
  (void)n_in; (void)out_size; (void)d_ws; (void)ws_size;
  const int*   features = (const int*)d_in[0];
  const int*   edge_src = (const int*)d_in[1];
  const int*   edge_dst = (const int*)d_in[2];
  const float* emb      = (const float*)d_in[3];
  const float* W        = (const float*)d_in[4];
  const float* bias     = (const float*)d_in[5];
  const float* gamma    = (const float*)d_in[6];
  const float* beta     = (const float*)d_in[7];
  float* out = (float*)d_out;

  int nGraphs = in_sizes[0] / kNPG;               // features is [B, 9]
  int grid = (nGraphs + kGPB - 1) / kGPB;         // 2048 blocks for B=32768

  // Allow >64KB dynamic LDS (gfx1250 WGP has 320KB).
  (void)hipFuncSetAttribute((const void*)gcn_fused_kernel,
                            hipFuncAttributeMaxDynamicSharedMemorySize, (int)kSmem);
  gcn_fused_kernel<<<grid, kThreads, kSmem, stream>>>(
      features, edge_src, edge_dst, emb, W, bias, gamma, beta, out, nGraphs);
}
